// RNNLayers_12859132084908
// MI455X (gfx1250) — matrix operand order repaired and checked by reading, not compiled, not run on hardware
//
#include <hip/hip_runtime.h>

// Problem constants (from reference): B=64, T=2048, F=64, H=128, 4H=512
#define B_ 64
#define T_ 2048
#define F_ 64
#define H_ 128
#define G_ 512

typedef __attribute__((ext_vector_type(16))) _Float16 v16h;
typedef __attribute__((ext_vector_type(8)))  _Float16 h8;
typedef __attribute__((ext_vector_type(8)))  float    v8f;
typedef unsigned int v4u __attribute__((ext_vector_type(4)));
typedef int          v8i __attribute__((ext_vector_type(8)));
typedef int          v4i __attribute__((ext_vector_type(4)));

// LDS row strides (in halfs)
#define XS  72    // x tile row: F + 8 (padded by TDM hardware)
#define WS  72    // W1^T row: F + 8
#define US  136   // U^T / W2^T row: H + 8
#define HS1 136   // layer-1 h buffer row (padded, lane-written)
#define HS2 128   // layer-2 h buffers row (unpadded: LDS budget = exactly 320KB)

__device__ __forceinline__ float sig_(float x){ return 1.0f/(1.0f+__expf(-x)); }
__device__ __forceinline__ float selu_(float x){
  const float a=1.6732632423543772f, l=1.0507009873554805f;
  return x > 0.0f ? l*x : l*a*(__expf(x)-1.0f);
}

// A-fragment (16x32 f16, MxK) from row-major LDS tile.
__device__ __forceinline__ v16h ldA(const _Float16* base, int stride, int m0, int k0, int lane){
  int row = m0 + (lane & 15);
  int ks  = k0 + ((lane >> 4) << 3);
  const h8* p0 = (const h8*)(base + row*stride + ks);
  const h8* p1 = (const h8*)(base + row*stride + ks + 16);
  h8 a0 = *p0, a1 = *p1;
  v16h r;
  #pragma unroll
  for (int j = 0; j < 8; ++j){ r[j] = a0[j]; r[j+8] = a1[j]; }
  return r;
}

// B-fragment (32x16 f16, KxN) from TRANSPOSED weights Wt[n][k] in LDS.
__device__ __forceinline__ v16h ldB(const _Float16* base, int stride, int n0, int k0, int lane){
  int row = n0 + (lane & 15);
  int ks  = k0 + ((lane >> 4) << 3);
  const h8* p0 = (const h8*)(base + row*stride + ks);
  const h8* p1 = (const h8*)(base + row*stride + ks + 16);
  h8 b0 = *p0, b1 = *p1;
  v16h r;
  #pragma unroll
  for (int j = 0; j < 8; ++j){ r[j] = b0[j]; r[j+8] = b1[j]; }
  return r;
}

#define WMMA(a,b,c) __builtin_amdgcn_wmma_f32_16x16x32_f16(false,(a),false,(b),(short)0,(c),false,false)

// ------------------------------------------------------------------
// Tensor Data Mover: 1-D global->LDS DMA with optional hardware LDS
// row padding (6-arg builtin form on this toolchain).
// ------------------------------------------------------------------
__device__ __forceinline__ void tdm_load_1d(unsigned lds_off, const void* gsrc,
                                            unsigned units8,
                                            unsigned pad_en, unsigned pad_iv, unsigned pad_amt){
  unsigned long long ga = (unsigned long long)(size_t)gsrc;
  v4u g0;
  g0[0] = 1u;                                            // count=1, user mode
  g0[1] = lds_off;                                       // LDS byte address
  g0[2] = (unsigned)ga;                                  // global addr lo
  g0[3] = ((unsigned)(ga >> 32) & 0x01ffffffu) | (2u << 30);   // addr hi | type=2
  v8i g1;
  g1[0] = (int)((3u << 16) | (pad_en << 20) | (pad_iv << 22) | (pad_amt << 25)); // 8B elems
  g1[1] = (int)((units8 & 0xffffu) << 16);               // tensor_dim0 lo16 (bits 63:48)
  g1[2] = (int)(((units8 >> 16) & 0xffffu) | (1u << 16));// tensor_dim0 hi16 | tensor_dim1=1
  g1[3] = (int)((units8 & 0xffffu) << 16);               // tile_dim0 (bits 127:112)
  g1[4] = 0;                                             // tile_dim1/2 unused
  g1[5] = (int)units8;                                   // tensor_dim0_stride lo32
  g1[6] = 0;
  g1[7] = 0;
  v4i z4 = (v4i)0;
  v8i z8 = (v8i)0;
  __builtin_amdgcn_tensor_load_to_lds(g0, g1, z4, z4, z8, 0);
}

// ------------------------------------------------------------------
// Kernel 0: x [B][T][F] f32  ->  x16 [T][B][F] f16 (timestep-contiguous)
// ------------------------------------------------------------------
__global__ void xcvt_kernel(const float* __restrict__ x, _Float16* __restrict__ x16){
  size_t i = (size_t)blockIdx.x * blockDim.x + threadIdx.x;   // over T*B*F
  int f = (int)(i % F_);
  size_t tb = i / F_;
  int b = (int)(tb % B_);
  size_t t = tb / B_;
  x16[i] = (_Float16)x[((size_t)b * T_ + t) * F_ + f];
}

// ------------------------------------------------------------------
// Layer 1: persistent single-WGP recurrent scan, TDM-streamed x tiles.
// k-block-major MAC order; zero-init accumulators (inline-0 C), bias
// folded into the gate nonlinearity.
// ------------------------------------------------------------------
__global__ __launch_bounds__(1024, 1)
void lstm1_kernel(const _Float16* __restrict__ x16,
                  const float* __restrict__ W1,
                  const float* __restrict__ U1,
                  const float* __restrict__ b1,
                  _Float16* __restrict__ h1g){
  extern __shared__ _Float16 sm1[];
  _Float16* W1t  = sm1;                 // [G_][WS]
  _Float16* U1t  = W1t + G_ * WS;       // [G_][US]
  _Float16* hbuf = U1t + G_ * US;       // [B_][HS1]
  _Float16* xb0  = hbuf + B_ * HS1;     // [B_][XS] double-buffered
  _Float16* xb1  = xb0 + B_ * XS;

  const int tid  = threadIdx.x;
  const int lane = tid & 31;
  const int wv   = tid >> 5;           // 0..31 (wave32)
  const int cg   = wv & 7;             // h-column group
  const int m0   = (wv >> 3) << 4;     // batch-row tile base
  const int nb   = cg * 16;            // base N column within a gate block
  const int nc   = nb + (lane & 15);

  for (int e = tid; e < F_ * G_; e += 1024){
    int r = e >> 9, c = e & 511;
    W1t[c * WS + r] = (_Float16)W1[e];
  }
  for (int e = tid; e < H_ * G_; e += 1024){
    int r = e >> 9, c = e & 511;
    U1t[c * US + r] = (_Float16)U1[e];
  }
  for (int e = tid; e < B_ * HS1; e += 1024) hbuf[e] = (_Float16)0.0f;

  float bias[4];
  #pragma unroll
  for (int g = 0; g < 4; ++g) bias[g] = b1[g * H_ + nc];

  v8f cst;
  #pragma unroll
  for (int j = 0; j < 8; ++j) cst[j] = 0.0f;

  const unsigned xoff0 = (unsigned)((const char*)xb0 - (const char*)sm1)
                         + __builtin_amdgcn_groupstaticsize();
  const unsigned xoff1 = xoff0 + (unsigned)(B_ * XS * sizeof(_Float16));
  const unsigned xu8   = (B_ * F_ * sizeof(_Float16)) / 8;  // 1024 8B units
  // pad: rows of 64 halfs = 32 DW -> interval code 4; +4 DW pad -> code 3
  if (wv == 0)
    tdm_load_1d(xoff0, x16, xu8, 1u, 4u, 3u);

  __syncthreads();

  for (int t = 0; t < T_; ++t){
    if (wv == 0){
      int tn = (t + 1 < T_) ? (t + 1) : t;                 // dummy re-issue at last step
      tdm_load_1d(((t + 1) & 1) ? xoff1 : xoff0,
                  x16 + (size_t)tn * (B_ * F_), xu8, 1u, 4u, 3u);
      __builtin_amdgcn_s_wait_tensorcnt(1);                // tile for step t is in LDS
    }
    __syncthreads();

    const _Float16* xb = (t & 1) ? xb1 : xb0;

    v8f acc[4];
    #pragma unroll
    for (int g = 0; g < 4; ++g){
      #pragma unroll
      for (int j = 0; j < 8; ++j) acc[g][j] = 0.0f;
    }

    // x_t @ W1 : K = 64 (one A-fragment live per k-block)
    #pragma unroll
    for (int kk = 0; kk < 2; ++kk){
      v16h a = ldA(xb, XS, m0, kk * 32, lane);
      #pragma unroll
      for (int g = 0; g < 4; ++g)
        acc[g] = WMMA(a, ldB(W1t, WS, g * H_ + nb, kk * 32, lane), acc[g]);
    }
    // h @ U1 : K = 128
    #pragma unroll
    for (int kk = 0; kk < 4; ++kk){
      v16h a = ldA(hbuf, HS1, m0, kk * 32, lane);
      #pragma unroll
      for (int g = 0; g < 4; ++g)
        acc[g] = WMMA(a, ldB(U1t, US, g * H_ + nb, kk * 32, lane), acc[g]);
    }
    __syncthreads();   // all waves done reading hbuf/xb for step t

    #pragma unroll
    for (int r = 0; r < 8; ++r){
      float ig = sig_(acc[0][r] + bias[0]);
      float fg = sig_(acc[1][r] + bias[1]);
      float gg = selu_(acc[2][r] + bias[2]);
      float og = sig_(acc[3][r] + bias[3]);
      float cn = fg * cst[r] + ig * gg;
      cst[r] = cn;
      float hv = og * selu_(cn);
      int rowB = m0 + r + ((lane >> 4) << 3);
      _Float16 hh = (_Float16)hv;
      hbuf[rowB * HS1 + nc] = hh;
      h1g[(size_t)t * (B_ * H_) + rowB * H_ + nc] = hh;
    }
    // next iteration's pre-compute barrier orders hbuf writes before reads
  }
}

// ------------------------------------------------------------------
// Layer 2: TDM-streamed (precomputed) h1 tiles; f32 output [B][T][H].
// ------------------------------------------------------------------
__global__ __launch_bounds__(1024, 1)
void lstm2_kernel(const _Float16* __restrict__ h1g,
                  const float* __restrict__ W2,
                  const float* __restrict__ U2,
                  const float* __restrict__ b2,
                  float* __restrict__ out){
  extern __shared__ _Float16 sm2[];
  _Float16* W2t   = sm2;                 // [G_][US]
  _Float16* U2t   = W2t + G_ * US;       // [G_][US]
  _Float16* h2buf = U2t + G_ * US;       // [B_][HS2]
  _Float16* hb0   = h2buf + B_ * HS2;    // [B_][HS2] double-buffered h1 tile
  _Float16* hb1   = hb0 + B_ * HS2;      // total = exactly 320 KB

  const int tid  = threadIdx.x;
  const int lane = tid & 31;
  const int wv   = tid >> 5;
  const int cg   = wv & 7;
  const int m0   = (wv >> 3) << 4;
  const int nb   = cg * 16;
  const int nc   = nb + (lane & 15);

  for (int e = tid; e < H_ * G_; e += 1024){
    int r = e >> 9, c = e & 511;
    W2t[c * US + r] = (_Float16)W2[e];
  }
  for (int e = tid; e < H_ * G_; e += 1024){
    int r = e >> 9, c = e & 511;
    U2t[c * US + r] = (_Float16)U2[e];
  }
  for (int e = tid; e < B_ * HS2; e += 1024) h2buf[e] = (_Float16)0.0f;

  float bias[4];
  #pragma unroll
  for (int g = 0; g < 4; ++g) bias[g] = b2[g * H_ + nc];

  v8f cst;
  #pragma unroll
  for (int j = 0; j < 8; ++j) cst[j] = 0.0f;

  const unsigned hoff0 = (unsigned)((const char*)hb0 - (const char*)sm2)
                         + __builtin_amdgcn_groupstaticsize();
  const unsigned hoff1 = hoff0 + (unsigned)(B_ * HS2 * sizeof(_Float16));
  const unsigned hu8   = (B_ * H_ * sizeof(_Float16)) / 8;   // 2048 8B units
  if (wv == 0)
    tdm_load_1d(hoff0, h1g, hu8, 0u, 0u, 0u);                // flat 16KB copy

  __syncthreads();

  for (int t = 0; t < T_; ++t){
    if (wv == 0){
      int tn = (t + 1 < T_) ? (t + 1) : t;
      tdm_load_1d(((t + 1) & 1) ? hoff1 : hoff0,
                  h1g + (size_t)tn * (B_ * H_), hu8, 0u, 0u, 0u);
      __builtin_amdgcn_s_wait_tensorcnt(1);
    }
    __syncthreads();

    const _Float16* h1b = (t & 1) ? hb1 : hb0;

    v8f acc[4];
    #pragma unroll
    for (int g = 0; g < 4; ++g){
      #pragma unroll
      for (int j = 0; j < 8; ++j) acc[g][j] = 0.0f;
    }

    // h1_t @ W2 : K = 128
    #pragma unroll
    for (int kk = 0; kk < 4; ++kk){
      v16h a = ldA(h1b, HS2, m0, kk * 32, lane);
      #pragma unroll
      for (int g = 0; g < 4; ++g)
        acc[g] = WMMA(a, ldB(W2t, US, g * H_ + nb, kk * 32, lane), acc[g]);
    }
    // h2 @ U2 : K = 128
    #pragma unroll
    for (int kk = 0; kk < 4; ++kk){
      v16h a = ldA(h2buf, HS2, m0, kk * 32, lane);
      #pragma unroll
      for (int g = 0; g < 4; ++g)
        acc[g] = WMMA(a, ldB(U2t, US, g * H_ + nb, kk * 32, lane), acc[g]);
    }
    __syncthreads();

    #pragma unroll
    for (int r = 0; r < 8; ++r){
      float ig = sig_(acc[0][r] + bias[0]);
      float fg = sig_(acc[1][r] + bias[1]);
      float gg = selu_(acc[2][r] + bias[2]);
      float og = sig_(acc[3][r] + bias[3]);
      float cn = fg * cst[r] + ig * gg;
      cst[r] = cn;
      float hv = og * selu_(cn);
      int rowB = m0 + r + ((lane >> 4) << 3);
      h2buf[rowB * HS2 + nc] = (_Float16)hv;
      out[(size_t)rowB * (T_ * H_) + (size_t)t * H_ + nc] = hv;
    }
  }
}

// ------------------------------------------------------------------
extern "C" void kernel_launch(void* const* d_in, const int* in_sizes, int n_in,
                              void* d_out, int out_size, void* d_ws, size_t ws_size,
                              hipStream_t stream) {
  const float* x  = (const float*)d_in[0];
  const float* W1 = (const float*)d_in[1];
  const float* U1 = (const float*)d_in[2];
  const float* b1 = (const float*)d_in[3];
  const float* W2 = (const float*)d_in[4];
  const float* U2 = (const float*)d_in[5];
  const float* b2 = (const float*)d_in[6];
  float* out = (float*)d_out;

  // ws layout: x16 [T][B][F] f16 (16 MB), then h1 [T][B][H] f16 (32 MB)
  _Float16* x16 = (_Float16*)d_ws;
  _Float16* h1g = x16 + (size_t)T_ * B_ * F_;

  size_t nx = (size_t)T_ * B_ * F_;
  xcvt_kernel<<<(int)(nx / 256), 256, 0, stream>>>(x, x16);

  size_t smem1 = (size_t)(G_ * WS + G_ * US + B_ * HS1 + 2 * B_ * XS) * sizeof(_Float16);
  lstm1_kernel<<<1, 1024, smem1, stream>>>(x16, W1, U1, b1, h1g);

  size_t smem2 = (size_t)(2 * G_ * US + 3 * B_ * HS2) * sizeof(_Float16);
  lstm2_kernel<<<1, 1024, smem2, stream>>>(h1g, W2, U2, b2, out);
}
